// SCAN_29420525977651
// MI455X (gfx1250) — matrix operand order repaired and checked by reading, not compiled, not run on hardware
//
#include <hip/hip_runtime.h>
#include <hip/hip_bf16.h>

typedef __attribute__((ext_vector_type(4)))  float  v4f;
typedef __attribute__((ext_vector_type(8)))  float  v8f;
typedef __attribute__((ext_vector_type(8)))  __bf16 v8bf;
typedef __attribute__((ext_vector_type(16))) __bf16 v16bf;

#define TN 1024  // output embedding width for both projections

__device__ __forceinline__ v8f wmma_bf16(v16bf a, v16bf b, v8f c) {
  // D = A(16x32 bf16) x B(32x16 bf16) + C(16x16 f32)
  return __builtin_amdgcn_wmma_f32_16x16x32_bf16(false, a, false, b, (short)0, c, false, false);
}

__device__ __forceinline__ void splitf(float f, __bf16& hi, __bf16& lo) {
  __bf16 h = (__bf16)f;           // RNE truncation to bf16
  hi = h;
  lo = (__bf16)(f - (float)h);    // residual
}

// ---------------------------------------------------------------------------
// C[M,1024] = A[M,K] @ W[K,1024] + bias, fp32 in/out, bf16x3 split WMMA.
// Block: 256 thr (8 waves). Block tile 128x128, wave tile 64x32 (4x2 WMMA).
// ---------------------------------------------------------------------------
__global__ __launch_bounds__(256) void gemm_bias_bf16x3(
    const float* __restrict__ A, const float* __restrict__ W,
    const float* __restrict__ bias, float* __restrict__ C,
    int M, int K)
{
  constexpr int BM = 128, BN = 128, BK = 32;
  __shared__ alignas(32) __bf16 sAhi[BM][BK];   // [m][k]
  __shared__ alignas(32) __bf16 sAlo[BM][BK];
  __shared__ alignas(32) __bf16 sBhi[BN][BK];   // transposed: [n][k]
  __shared__ alignas(32) __bf16 sBlo[BN][BK];

  const int tid  = threadIdx.x;
  const int lane = tid & 31;
  const int wave = tid >> 5;
  const int h    = lane >> 4;     // wave half (0/1)
  const int ln   = lane & 15;
  const int wm   = wave >> 2;     // 0..1  -> 64 rows
  const int wn   = wave & 3;      // 0..3  -> 32 cols
  const int m0   = blockIdx.y * BM;
  const int n0   = blockIdx.x * BN;

  v8f acc[4][2];
  #pragma unroll
  for (int i = 0; i < 4; ++i)
    #pragma unroll
    for (int j = 0; j < 2; ++j) acc[i][j] = (v8f){};

  const int ksteps = (K + BK - 1) / BK;
  for (int ks = 0; ks < ksteps; ++ks) {
    const int k0 = ks * BK;

    // ---- stage global tiles into registers (coalesced float4) -------------
    v4f aval[4], wval[4];
    #pragma unroll
    for (int j = 0; j < 4; ++j) {
      const int q  = j * 256 + tid;          // 0..1023 chunks of A tile
      const int r  = q >> 3;                 // row in tile 0..127
      const int c4 = q & 7;                  // float4 along k
      const float* p = A + (size_t)(m0 + r) * K + (k0 + c4 * 4);
      if (k0 + BK <= K) {
        aval[j] = *(const v4f*)p;
      } else {                               // K tail (K=300 case): zero pad
        v4f t = {};
        #pragma unroll
        for (int e = 0; e < 4; ++e) { int kk = k0 + c4 * 4 + e; if (kk < K) t[e] = p[e]; }
        aval[j] = t;
      }
    }
    #pragma unroll
    for (int j = 0; j < 4; ++j) {
      const int q  = j * 256 + tid;          // chunks of W tile (32 x 128)
      const int kk = q >> 5;                 // k in tile 0..31
      const int c4 = q & 31;                 // float4 along n
      if (k0 + kk < K) wval[j] = *(const v4f*)(W + (size_t)(k0 + kk) * TN + (n0 + c4 * 4));
      else             wval[j] = (v4f){};
    }
    if (k0 + BK < K) {                       // hint next K tile toward L2/L0
      __builtin_prefetch(A + (size_t)(m0 + (tid >> 3)) * K + (k0 + BK), 0, 1);
      __builtin_prefetch(W + (size_t)(k0 + BK + (tid >> 5)) * TN + n0 + (tid & 31) * 4, 0, 1);
    }

    __syncthreads();                         // prev-iter LDS reads complete
    // ---- split fp32 -> bf16 hi/lo while storing to LDS --------------------
    #pragma unroll
    for (int j = 0; j < 4; ++j) {
      const int q  = j * 256 + tid;
      const int r  = q >> 3;
      const int c4 = q & 7;
      #pragma unroll
      for (int e = 0; e < 4; ++e) splitf(aval[j][e], sAhi[r][c4 * 4 + e], sAlo[r][c4 * 4 + e]);
    }
    #pragma unroll
    for (int j = 0; j < 4; ++j) {
      const int q  = j * 256 + tid;
      const int kk = q >> 5;
      const int c4 = q & 31;
      #pragma unroll
      for (int e = 0; e < 4; ++e) splitf(wval[j][e], sBhi[c4 * 4 + e][kk], sBlo[c4 * 4 + e][kk]);
    }
    __syncthreads();

    // ---- build WMMA fragments from LDS ------------------------------------
    // A 16-bit layout: lane ln row M; element i -> K = h*16 + i  (k-contiguous)
    // B 16-bit layout: lane ln col N; element i -> K = h*16 + i  (k-contiguous)
    v16bf afh[4], afl[4], bfh[2], bfl[2];
    #pragma unroll
    for (int mt = 0; mt < 4; ++mt) {
      const int r = wm * 64 + mt * 16 + ln;
      v8bf a0 = *(const v8bf*)&sAhi[r][h * 16];
      v8bf a1 = *(const v8bf*)&sAhi[r][h * 16 + 8];
      afh[mt] = __builtin_shufflevector(a0, a1, 0,1,2,3,4,5,6,7,8,9,10,11,12,13,14,15);
      v8bf l0 = *(const v8bf*)&sAlo[r][h * 16];
      v8bf l1 = *(const v8bf*)&sAlo[r][h * 16 + 8];
      afl[mt] = __builtin_shufflevector(l0, l1, 0,1,2,3,4,5,6,7,8,9,10,11,12,13,14,15);
    }
    #pragma unroll
    for (int nt = 0; nt < 2; ++nt) {
      const int n = wn * 32 + nt * 16 + ln;
      bfh[nt] = *(const v16bf*)&sBhi[n][h * 16];
      bfl[nt] = *(const v16bf*)&sBlo[n][h * 16];
    }
    // ---- 4x2 tiles x 3 split-products -------------------------------------
    #pragma unroll
    for (int mt = 0; mt < 4; ++mt)
      #pragma unroll
      for (int nt = 0; nt < 2; ++nt) {
        acc[mt][nt] = wmma_bf16(afh[mt], bfh[nt], acc[mt][nt]);
        acc[mt][nt] = wmma_bf16(afh[mt], bfl[nt], acc[mt][nt]);
        acc[mt][nt] = wmma_bf16(afl[mt], bfh[nt], acc[mt][nt]);
      }
  }

  // ---- epilogue: bias add, store fp32. C layout: vgpr v -> row v + h*8 ----
  #pragma unroll
  for (int nt = 0; nt < 2; ++nt) {
    const int col = n0 + wn * 32 + nt * 16 + ln;
    const float bv = bias[col];
    #pragma unroll
    for (int mt = 0; mt < 4; ++mt) {
      const int rbase = m0 + wm * 64 + mt * 16 + h * 8;
      #pragma unroll
      for (int v = 0; v < 8; ++v)
        C[(size_t)(rbase + v) * TN + col] = acc[mt][nt][v] + bv;
    }
  }
}

// ---------------------------------------------------------------------------
// Row-wise l2norm (D=1024): one wave per row, 8 rows per block.
// ---------------------------------------------------------------------------
__global__ __launch_bounds__(256) void l2norm_rows(float* __restrict__ X, int rows)
{
  const int wave = threadIdx.x >> 5;
  const int lane = threadIdx.x & 31;
  const int row  = blockIdx.x * 8 + wave;
  if (row >= rows) return;
  float* p = X + (size_t)row * 1024;
  v4f v[8];
  float s = 0.f;
  #pragma unroll
  for (int j = 0; j < 8; ++j) {
    v[j] = *(const v4f*)(p + lane * 4 + j * 128);
    s += v[j][0]*v[j][0] + v[j][1]*v[j][1] + v[j][2]*v[j][2] + v[j][3]*v[j][3];
  }
  #pragma unroll
  for (int off = 16; off > 0; off >>= 1) s += __shfl_xor(s, off, 32);
  const float sc = rsqrtf(s + 1e-12f);
  #pragma unroll
  for (int j = 0; j < 8; ++j) *(v4f*)(p + lane * 4 + j * 128) = v[j] * sc;
}

// ---------------------------------------------------------------------------
// Per-batch fused attention score. One block (8 waves) per batch.
// P = [cap(60); img(36)] @ img^T  computed with split-bf16 WMMA:
//   P[0..59][r]  = sim,  P[60+r][r'] = Gram G = img.img^T
// Then per word: softmax(9*sim); num = sum w*sim; ||att||^2 = w^T G w.
// ---------------------------------------------------------------------------
__global__ __launch_bounds__(256) void scan_attn_score(
    const float* __restrict__ cap, const float* __restrict__ img,
    const int* __restrict__ caplen, float* __restrict__ out)
{
  const int b = blockIdx.x;
  __shared__ float P[96][48];
  __shared__ float Wsh[60][37];
  __shared__ float rsh[60];
  const int tid  = threadIdx.x;
  const int lane = tid & 31, wave = tid >> 5;
  const int h = lane >> 4, ln = lane & 15;
  const float* capb = cap + (size_t)b * 60 * 1024;
  const float* imgb = img + (size_t)b * 36 * 1024;

  // 6 M-tiles (96 rows: 60 cap + 36 img, no pad) x 3 N-tiles (36 cols pad 48)
  for (int t = wave; t < 18; t += 8) {
    const int mt = t / 3, nt = t % 3;
    const int grow = mt * 16 + ln;
    const float* arow = (grow < 60) ? capb + (size_t)grow * 1024
                                    : imgb + (size_t)(grow - 60) * 1024;
    const int n = nt * 16 + ln;
    const float* brow = imgb + (size_t)(n < 36 ? n : 35) * 1024;  // pad cols clamp
    v8f acc = {};
    for (int k0 = 0; k0 < 1024; k0 += 32) {
      v16bf ah, al, bh, bl;
      const float* p0 = arow + k0 + h * 16;
      const float* p1 = arow + k0 + h * 16 + 8;
      #pragma unroll
      for (int i = 0; i < 8; ++i) {
        __bf16 hi, lo;
        splitf(p0[i], hi, lo); ah[i]     = hi; al[i]     = lo;
        splitf(p1[i], hi, lo); ah[i + 8] = hi; al[i + 8] = lo;
      }
      const float* pb = brow + k0 + h * 16;
      #pragma unroll
      for (int i = 0; i < 16; ++i) {
        __bf16 hi, lo; splitf(pb[i], hi, lo); bh[i] = hi; bl[i] = lo;
      }
      acc = wmma_bf16(ah, bh, acc);
      acc = wmma_bf16(ah, bl, acc);
      acc = wmma_bf16(al, bh, acc);
    }
    #pragma unroll
    for (int v = 0; v < 8; ++v) P[mt * 16 + h * 8 + v][nt * 16 + ln] = acc[v];
  }
  __syncthreads();

  if (tid < 60) {
    float mx = -1e30f;
    for (int r = 0; r < 36; ++r) mx = fmaxf(mx, P[tid][r]);
    float s = 0.f;
    for (int r = 0; r < 36; ++r) {
      float e = __expf(9.0f * (P[tid][r] - mx));
      Wsh[tid][r] = e; s += e;
    }
    const float inv = 1.0f / s;
    float num = 0.f, att2 = 0.f;
    for (int r = 0; r < 36; ++r) {
      const float wr = Wsh[tid][r];
      num += wr * P[tid][r];                 // w . sim
      float tr = 0.f;
      for (int r2 = 0; r2 < 36; ++r2) tr += Wsh[tid][r2] * P[60 + r][r2];
      att2 += wr * tr;                       // w^T G w
    }
    num  *= inv;
    att2 *= inv * inv;
    const float attn = sqrtf(fmaxf(att2, 0.f));
    const float rl = num / fmaxf(attn, 1e-8f);   // ||cap||==1 after l2norm
    rsh[tid] = (tid < caplen[b]) ? rl : 0.f;
  }
  __syncthreads();
  if (tid == 0) {
    float s = 0.f;
    for (int l = 0; l < 60; ++l) s += rsh[l];
    out[b] = s / (float)caplen[b];
  }
}

// ---------------------------------------------------------------------------
extern "C" void kernel_launch(void* const* d_in, const int* in_sizes, int n_in,
                              void* d_out, int out_size, void* d_ws, size_t ws_size,
                              hipStream_t stream)
{
  (void)in_sizes; (void)n_in; (void)out_size; (void)ws_size;
  const float* images   = (const float*)d_in[0];  // 256 x 36 x 2048
  const float* captions = (const float*)d_in[1];  // 256 x 60 x 300
  const int*   caplen   = (const int*)  d_in[2];  // 256
  const float* W_img    = (const float*)d_in[3];  // 2048 x 1024
  const float* b_img    = (const float*)d_in[4];  // 1024
  const float* W_txt    = (const float*)d_in[5];  // 300 x 1024
  const float* b_txt    = (const float*)d_in[6];  // 1024
  float* out = (float*)d_out;                     // 256

  float* img_embs = (float*)d_ws;                     //  9216 x 1024 fp32
  float* cap_embs = img_embs + (size_t)9216 * 1024;   // 15360 x 1024 fp32

  gemm_bias_bf16x3<<<dim3(8, 72),  256, 0, stream>>>(images,   W_img, b_img, img_embs,  9216, 2048);
  gemm_bias_bf16x3<<<dim3(8, 120), 256, 0, stream>>>(captions, W_txt, b_txt, cap_embs, 15360,  300);
  l2norm_rows<<<9216 / 8,  256, 0, stream>>>(img_embs,  9216);
  l2norm_rows<<<15360 / 8, 256, 0, stream>>>(cap_embs, 15360);
  scan_attn_score<<<256, 256, 0, stream>>>(cap_embs, img_embs, caplen, out);
}
